// MultiHeadAttentionLayer_19585050870068
// MI455X (gfx1250) — matrix operand order repaired and checked by reading, compiled, not run on hardware
//
#include <hip/hip_runtime.h>

// ---------------------------------------------------------------------------
// Types for CDNA5 WMMA
// ---------------------------------------------------------------------------
typedef float          v8f    __attribute__((ext_vector_type(8)));
typedef __bf16         v16bf  __attribute__((ext_vector_type(16)));
typedef unsigned short v16u16 __attribute__((ext_vector_type(16)));
typedef unsigned short v8u16  __attribute__((ext_vector_type(8)));

#define NN   50000
#define DD   128
#define EE   1600000
#define HH   8
#define DHH  16
#define FFH  512

__device__ __forceinline__ unsigned short f32_to_bf16(float f) {
    unsigned int u = __float_as_uint(f);
    u += 0x7FFFu + ((u >> 16) & 1u);           // round-to-nearest-even
    return (unsigned short)(u >> 16);
}

__device__ __forceinline__ void atomicMaxF(float* addr, float val) {
    unsigned int* ua = (unsigned int*)addr;
    unsigned int old = *ua;
    while (true) {
        float f = __uint_as_float(old);
        if (f >= val) break;
        unsigned int assumed = old;
        old = atomicCAS(ua, assumed, __float_as_uint(val));
        if (old == assumed) break;
    }
}

// ---------------------------------------------------------------------------
// Elementwise helpers
// ---------------------------------------------------------------------------
__global__ void cast_bf16_kernel(const float* __restrict__ in,
                                 unsigned short* __restrict__ out, int n) {
    int i = blockIdx.x * blockDim.x + threadIdx.x;
    if (i < n) out[i] = f32_to_bf16(in[i]);
}

__global__ void fill_f32_kernel(float* __restrict__ p, float v, int n) {
    int i = blockIdx.x * blockDim.x + threadIdx.x;
    if (i < n) p[i] = v;
}

// ---------------------------------------------------------------------------
// bf16 WMMA GEMM, LDS-staged:  C[M x Nc] = A[M x K] * B[K x Nc], f32 acc.
//  - A strip (16 rows x K) staged into LDS via GLOBAL_LOAD_ASYNC_TO_LDS_B128
//    (ASYNCcnt-tracked), rows padded +8 elements to de-conflict banks.
//  - B panel staged transposed (Bt[Nc][K], padded) so each lane's 16 K-values
//    are two 16B LDS reads, 16 elements apart (2x ds_load_b128) per WMMA.
//  - One wave per 16x16 C tile; 8 waves/block cover a 16-row C strip.
//  MODE 0: f32 store to Cf.  MODE 1: bias + ReLU + bf16 store to Cb.
// VGPR layouts per CDNA5 ISA 7.12.2 (wave32):
//   A 16x32 bf16 : m=lane&15, hi=lane>>4 -> k = j + (j&8) + 8*hi
//   B 32x16 bf16 : n=lane&15, k = j + (j&8) + 8*hi (same as A)
//   C 16x16 f32  : row = 8*hi + r, col = lane&15
// ---------------------------------------------------------------------------
template <int K, int Nc, int MODE>
__global__ __launch_bounds__(256) void gemm_bf16_wmma(
    const unsigned short* __restrict__ A, const unsigned short* __restrict__ B,
    float* __restrict__ Cf, unsigned short* __restrict__ Cb,
    const float* __restrict__ bias)
{
    constexpr int KP = K + 8;                 // padded row (16B) -> no bank conflicts
    extern __shared__ unsigned short smem[];
    unsigned short* As = smem;                // [16][KP]
    unsigned short* Bt = smem + 16 * KP;      // [Nc][KP]  (transposed B)

    const int tid  = threadIdx.x;
    const int row0 = blockIdx.x * 16;

    // ---- async-stage A strip: 2*K chunks of 16B, CDNA5 async-to-LDS path ----
    {
        constexpr int CPR = K / 8;            // 16B chunks per row
        #pragma unroll
        for (int c0 = 0; c0 < 2 * K; c0 += 256) {
            const int c  = c0 + tid;
            const int r  = c / CPR;
            const int cc = c - r * CPR;
            const unsigned ldsOff =
                (unsigned)(r * KP + cc * 8) * 2u;            // As base offset 0
            const unsigned long long ga =
                (unsigned long long)(A + (size_t)(row0 + r) * K + cc * 8);
            asm volatile("global_load_async_to_lds_b128 %0, %1, off"
                         :: "v"(ldsOff), "v"(ga) : "memory");
        }
    }
    // ---- stage B transposed (coalesced global reads, scattered LDS writes) ----
    {
        constexpr int TOT = K * Nc;
        for (int idx = tid; idx < TOT; idx += 256) {
            const int k = idx / Nc, n = idx - k * Nc;
            Bt[(size_t)n * KP + k] = B[idx];
        }
    }
    asm volatile("s_wait_asynccnt 0x0" ::: "memory");
    __syncthreads();

    const int wave = tid >> 5;
    const int lane = tid & 31;
    const int ml   = lane & 15;
    const int hi   = lane >> 4;

    for (int ct = wave; ct < (Nc >> 4); ct += 8) {
        const int n0 = ct << 4;
        v8f acc = {};
        const unsigned short* arow = As + ml * KP;
        const unsigned short* brow = Bt + (size_t)(n0 + ml) * KP;
        #pragma unroll
        for (int k0 = 0; k0 < K; k0 += 32) {
            v8u16 alo = *reinterpret_cast<const v8u16*>(arow + k0 + hi * 8);
            v8u16 ahi = *reinterpret_cast<const v8u16*>(arow + k0 + hi * 8 + 16);
            v16u16 a = __builtin_shufflevector(alo, ahi,
                0,1,2,3,4,5,6,7,8,9,10,11,12,13,14,15);
            v8u16 blo = *reinterpret_cast<const v8u16*>(brow + k0 + hi * 8);
            v8u16 bhi = *reinterpret_cast<const v8u16*>(brow + k0 + hi * 8 + 16);
            v16u16 b = __builtin_shufflevector(blo, bhi,
                0,1,2,3,4,5,6,7,8,9,10,11,12,13,14,15);
            acc = __builtin_amdgcn_wmma_f32_16x16x32_bf16(
                false, __builtin_bit_cast(v16bf, a),
                false, __builtin_bit_cast(v16bf, b),
                (short)0, acc, false, false);
        }
        const int col = n0 + ml;
        #pragma unroll
        for (int r = 0; r < 8; ++r) {
            const size_t idx = (size_t)(row0 + hi * 8 + r) * Nc + col;
            if (MODE == 0) {
                Cf[idx] = acc[r];
            } else {
                float t = acc[r] + bias[col];
                t = t > 0.0f ? t : 0.0f;
                Cb[idx] = f32_to_bf16(t);
            }
        }
    }
}

// ---------------------------------------------------------------------------
// Edge-parallel segment softmax + aggregation (scores cached in ws)
// ---------------------------------------------------------------------------
__global__ void edge_pass_a(const float* __restrict__ q, const float* __restrict__ k,
                            const long long* __restrict__ src, const long long* __restrict__ dst,
                            float* __restrict__ sc, float* __restrict__ mbuf, int E)
{
    int e = blockIdx.x * blockDim.x + threadIdx.x;
    if (e >= E) return;
    const size_t d = (size_t)dst[e], s = (size_t)src[e];
    const float* qr = q + d * DD;
    const float* kr = k + s * DD;
    #pragma unroll
    for (int h = 0; h < HH; ++h) {
        float acc = 0.0f;
        #pragma unroll
        for (int t = 0; t < DHH; ++t) acc += qr[h * DHH + t] * kr[h * DHH + t];
        acc *= 0.25f;                              // 1/sqrt(16)
        sc[(size_t)e * HH + h] = acc;
        atomicMaxF(&mbuf[d * HH + h], acc);
    }
}

__global__ void edge_pass_b(const long long* __restrict__ dst,
                            float* __restrict__ sc, const float* __restrict__ mbuf,
                            float* __restrict__ denom, int E)
{
    int e = blockIdx.x * blockDim.x + threadIdx.x;
    if (e >= E) return;
    const size_t d = (size_t)dst[e];
    #pragma unroll
    for (int h = 0; h < HH; ++h) {
        float ex = __expf(sc[(size_t)e * HH + h] - mbuf[d * HH + h]);
        sc[(size_t)e * HH + h] = ex;
        atomicAdd(&denom[d * HH + h], ex);
    }
}

__global__ void edge_pass_c(const float* __restrict__ v,
                            const long long* __restrict__ src, const long long* __restrict__ dst,
                            const float* __restrict__ sc, const float* __restrict__ denom,
                            float* __restrict__ agg, int E)
{
    int e = blockIdx.x * blockDim.x + threadIdx.x;
    if (e >= E) return;
    const size_t d = (size_t)dst[e], s = (size_t)src[e];
    #pragma unroll
    for (int h = 0; h < HH; ++h) {
        float alpha = sc[(size_t)e * HH + h] / (denom[d * HH + h] + 1e-9f);
        #pragma unroll
        for (int t = 0; t < DHH; ++t) {
            atomicAdd(&agg[d * DD + h * DHH + t],
                      alpha * v[s * DD + h * DHH + t]);
        }
    }
}

// ---------------------------------------------------------------------------
// out = LayerNorm(x + y [+ bias])  — one wave per 128-wide row, 8 rows/block
// ---------------------------------------------------------------------------
__global__ __launch_bounds__(256) void residual_ln_kernel(
    const float* __restrict__ x, const float* __restrict__ y,
    const float* __restrict__ bias,
    const float* __restrict__ g, const float* __restrict__ b,
    float* __restrict__ out, int N)
{
    int row = blockIdx.x * 8 + (threadIdx.x >> 5);
    if (row >= N) return;
    int lane = threadIdx.x & 31;
    const float* xr = x + (size_t)row * DD;
    const float* yr = y + (size_t)row * DD;
    float v[4], s = 0.0f, s2 = 0.0f;
    #pragma unroll
    for (int i = 0; i < 4; ++i) {
        int c = lane + i * 32;
        float t = xr[c] + yr[c];
        if (bias) t += bias[c];
        v[i] = t; s += t; s2 += t * t;
    }
    #pragma unroll
    for (int m = 16; m > 0; m >>= 1) {
        s  += __shfl_xor(s,  m, 32);
        s2 += __shfl_xor(s2, m, 32);
    }
    float mu  = s * (1.0f / DD);
    float var = s2 * (1.0f / DD) - mu * mu;
    float r   = rsqrtf(var + 1e-5f);
    #pragma unroll
    for (int i = 0; i < 4; ++i) {
        int c = lane + i * 32;
        out[(size_t)row * DD + c] = (v[i] - mu) * r * g[c] + b[c];
    }
}

// ---------------------------------------------------------------------------
// Host-side orchestration
// ---------------------------------------------------------------------------
extern "C" void kernel_launch(void* const* d_in, const int* in_sizes, int n_in,
                              void* d_out, int out_size, void* d_ws, size_t ws_size,
                              hipStream_t stream)
{
    const float*     X    = (const float*)d_in[0];
    const long long* ei   = (const long long*)d_in[1];
    const float*     Wq   = (const float*)d_in[2];
    const float*     Wk   = (const float*)d_in[3];
    const float*     Wv   = (const float*)d_in[4];
    const float*     W0   = (const float*)d_in[5];
    const float*     ln1g = (const float*)d_in[6];
    const float*     ln1b = (const float*)d_in[7];
    const float*     fc1w = (const float*)d_in[8];
    const float*     fc1b = (const float*)d_in[9];
    const float*     fc2w = (const float*)d_in[10];
    const float*     fc2b = (const float*)d_in[11];
    const float*     ln2g = (const float*)d_in[12];
    const float*     ln2b = (const float*)d_in[13];

    const long long* src = ei;            // edge_index[0]
    const long long* dst = ei + EE;       // edge_index[1]

    // --- workspace carve (256B aligned) ---
    char* base = (char*)d_ws;
    size_t off = 0;
    auto carve = [&](size_t bytes) -> char* {
        char* p = base + off;
        off += (bytes + 255) & ~(size_t)255;
        return p;
    };
    unsigned short* Xb   = (unsigned short*)carve((size_t)NN * DD * 2); // reused: aggb, h1b
    unsigned short* Wqb  = (unsigned short*)carve(DD * DD * 2);
    unsigned short* Wkb  = (unsigned short*)carve(DD * DD * 2);
    unsigned short* Wvb  = (unsigned short*)carve(DD * DD * 2);
    unsigned short* W0b  = (unsigned short*)carve(DD * DD * 2);
    unsigned short* F1b  = (unsigned short*)carve(DD * FFH * 2);
    unsigned short* F2b  = (unsigned short*)carve(FFH * DD * 2);
    float* qf    = (float*)carve((size_t)NN * DD * 4);   // reused: emb
    float* kf    = (float*)carve((size_t)NN * DD * 4);   // reused: h1
    float* vf    = (float*)carve((size_t)NN * DD * 4);
    float* mbuf  = (float*)carve((size_t)NN * HH * 4);
    float* dbuf  = (float*)carve((size_t)NN * HH * 4);
    float* agg   = (float*)carve((size_t)NN * DD * 4);   // reused: ff2
    float* sc    = (float*)carve((size_t)EE * HH * 4);
    unsigned short* ffb = (unsigned short*)carve((size_t)NN * FFH * 2);

    float* emb = qf;
    float* h1  = kf;
    float* ff2 = agg;
    unsigned short* aggb = Xb;
    unsigned short* h1b  = Xb;

    const int T = 256;
    auto blks = [](int n, int t) { return (n + t - 1) / t; };

    // dynamic LDS sizes: (16 + Nc) * (K + 8) * 2 bytes
    const size_t lds_d_d   = (size_t)(16 + DD)  * (DD  + 8) * 2;  //  ~39 KB
    const size_t lds_d_ff  = (size_t)(16 + FFH) * (DD  + 8) * 2;  // ~140 KB
    const size_t lds_ff_d  = (size_t)(16 + DD)  * (FFH + 8) * 2;  // ~146 KB

    // 1. bf16 casts of X and all weight matrices
    cast_bf16_kernel<<<blks(NN * DD, T), T, 0, stream>>>(X,    Xb,  NN * DD);
    cast_bf16_kernel<<<blks(DD * DD, T), T, 0, stream>>>(Wq,   Wqb, DD * DD);
    cast_bf16_kernel<<<blks(DD * DD, T), T, 0, stream>>>(Wk,   Wkb, DD * DD);
    cast_bf16_kernel<<<blks(DD * DD, T), T, 0, stream>>>(Wv,   Wvb, DD * DD);
    cast_bf16_kernel<<<blks(DD * DD, T), T, 0, stream>>>(W0,   W0b, DD * DD);
    cast_bf16_kernel<<<blks(DD * FFH, T), T, 0, stream>>>(fc1w, F1b, DD * FFH);
    cast_bf16_kernel<<<blks(FFH * DD, T), T, 0, stream>>>(fc2w, F2b, FFH * DD);

    // 2. QKV projections (WMMA bf16, f32 accumulate)
    const int gemmBlocks = NN / 16; // 3125
    gemm_bf16_wmma<DD, DD, 0><<<gemmBlocks, 256, lds_d_d, stream>>>(Xb, Wqb, qf, nullptr, nullptr);
    gemm_bf16_wmma<DD, DD, 0><<<gemmBlocks, 256, lds_d_d, stream>>>(Xb, Wkb, kf, nullptr, nullptr);
    gemm_bf16_wmma<DD, DD, 0><<<gemmBlocks, 256, lds_d_d, stream>>>(Xb, Wvb, vf, nullptr, nullptr);

    // 3. init segment buffers
    fill_f32_kernel<<<blks(NN * HH, T), T, 0, stream>>>(mbuf, -3.0e38f, NN * HH);
    fill_f32_kernel<<<blks(NN * HH, T), T, 0, stream>>>(dbuf, 0.0f,     NN * HH);
    fill_f32_kernel<<<blks(NN * DD, T), T, 0, stream>>>(agg,  0.0f,     NN * DD);

    // 4-6. edge-parallel segment softmax + weighted aggregation
    const int eBlocks = blks(EE, T);
    edge_pass_a<<<eBlocks, T, 0, stream>>>(qf, kf, src, dst, sc, mbuf, EE);
    edge_pass_b<<<eBlocks, T, 0, stream>>>(dst, sc, mbuf, dbuf, EE);
    edge_pass_c<<<eBlocks, T, 0, stream>>>(vf, src, dst, sc, dbuf, agg, EE);

    // 7-8. embedding = agg @ W0
    cast_bf16_kernel<<<blks(NN * DD, T), T, 0, stream>>>(agg, aggb, NN * DD);
    gemm_bf16_wmma<DD, DD, 0><<<gemmBlocks, 256, lds_d_d, stream>>>(aggb, W0b, emb, nullptr, nullptr);

    // 9. h1 = LN(X + embedding)
    residual_ln_kernel<<<blks(NN, 8), 256, 0, stream>>>(X, emb, nullptr, ln1g, ln1b, h1, NN);

    // 10-11. FFN up: relu(h1 @ fc1 + b1) -> bf16  (fused in GEMM epilogue)
    cast_bf16_kernel<<<blks(NN * DD, T), T, 0, stream>>>(h1, h1b, NN * DD);
    gemm_bf16_wmma<DD, FFH, 1><<<gemmBlocks, 256, lds_d_ff, stream>>>(h1b, F1b, nullptr, ffb, fc1b);

    // 12. FFN down: ff2 = ffb @ fc2   (fc2 bias folded into final LN kernel)
    gemm_bf16_wmma<FFH, DD, 0><<<gemmBlocks, 256, lds_ff_d, stream>>>(ffb, F2b, ff2, nullptr, nullptr);

    // 13. out = LN(h1 + ff2 + fc2_b)
    residual_ln_kernel<<<blks(NN, 8), 256, 0, stream>>>(h1, ff2, fc2b, ln2g, ln2b, (float*)d_out, NN);
}
